// TCAttention_35089882808567
// MI455X (gfx1250) — matrix operand order repaired and checked by reading, not compiled, not run on hardware
//
#include <hip/hip_runtime.h>
#include <hip/hip_bf16.h>
#include <stdint.h>

// Problem constants (from reference)
#define T_      2048
#define C_      4096
#define H_      32
#define DH_     128
#define FRAME_  576.0f
#define GAMMA_  1.0f
#define LOG2E_      1.4426950408889634f
#define LOG2_10000_ 13.287712379549449f
#define SCALE_      0.08838834764831845f   // 1/sqrt(128)

typedef __attribute__((ext_vector_type(16))) __bf16         v16bf;
typedef __attribute__((ext_vector_type(16))) unsigned short v16u;
typedef __attribute__((ext_vector_type(8)))  unsigned short v8u;
typedef __attribute__((ext_vector_type(8)))  float          v8f;

// Tensor Data Mover availability (device pass only; host pass falls back).
#if defined(__gfx1250__) && __has_builtin(__builtin_amdgcn_tensor_load_to_lds)
#define USE_TDM 1
#else
#define USE_TDM 0
#endif

__device__ inline unsigned short f2bf(float f) {
    unsigned int u = __float_as_uint(f);
    u += 0x7FFFu + ((u >> 16) & 1u);           // round-to-nearest-even
    return (unsigned short)(u >> 16);
}

__device__ inline v16bf bc16(v8u lo, v8u hi) {
    v16u w = __builtin_shufflevector(lo, hi, 0,1,2,3,4,5,6,7,8,9,10,11,12,13,14,15);
    return __builtin_bit_cast(v16bf, w);
}

__device__ inline v8f wmma_bf16(v16bf a, v16bf b, v8f c) {
    return __builtin_amdgcn_wmma_f32_16x16x32_bf16(false, a, false, b, (short)0, c, false, false);
}

__device__ inline float rmax16(float x) {
    #pragma unroll
    for (int m = 1; m < 16; m <<= 1) x = fmaxf(x, __shfl_xor(x, m, 32));
    return x;
}
__device__ inline float rsum16(float x) {
    #pragma unroll
    for (int m = 1; m < 16; m <<= 1) x += __shfl_xor(x, m, 32);
    return x;
}

// Load 16 consecutive values starting at p, converted to bf16.
__device__ inline void ld16bf(const float* __restrict__ p, v8u& lo, v8u& hi) {
    const float4* q = (const float4*)p;
    float4 f0 = q[0], f1 = q[1], f2 = q[2], f3 = q[3];
    v8u a, b;
    a[0]=f2bf(f0.x); a[1]=f2bf(f0.y); a[2]=f2bf(f0.z); a[3]=f2bf(f0.w);
    a[4]=f2bf(f1.x); a[5]=f2bf(f1.y); a[6]=f2bf(f1.z); a[7]=f2bf(f1.w);
    b[0]=f2bf(f2.x); b[1]=f2bf(f2.y); b[2]=f2bf(f2.z); b[3]=f2bf(f2.w);
    b[4]=f2bf(f3.x); b[5]=f2bf(f3.y); b[6]=f2bf(f3.z); b[7]=f2bf(f3.w);
    lo = a; hi = b;
}
__device__ inline void ld16bf(const unsigned short* __restrict__ p, v8u& lo, v8u& hi) {
    lo = ((const v8u*)p)[0];
    hi = ((const v8u*)p)[1];
}

#if USE_TDM
typedef __attribute__((ext_vector_type(4))) unsigned int tdm_g0_t;
typedef __attribute__((ext_vector_type(8))) int          tdm_g1_t;
typedef __attribute__((ext_vector_type(4))) int          tdm_g2_t;

// Issue a 2-D TDM tile load (bf16 elements) into LDS.
//   tile_w elements contiguous per row, tile_h rows, row stride `stride`
//   (all in elements). D# packing per CDNA5 ISA §8.3/§8.4.
__device__ inline void tdm_load_2d(uint32_t lds_off, const void* gptr,
                                   uint32_t tile_w, uint32_t tile_h,
                                   uint32_t stride, uint32_t tensor_w,
                                   uint32_t tensor_h) {
    uint64_t ga = (uint64_t)(uintptr_t)gptr;
    tdm_g0_t g0;
    g0[0] = 1u;                                   // count=1 (valid), user mode
    g0[1] = lds_off;                              // lds_addr (bytes)
    g0[2] = (uint32_t)ga;                         // global_addr[31:0]
    g0[3] = (uint32_t)((ga >> 32) & 0x01FFFFFFu)  // global_addr[56:32]
            | (2u << 30);                         // type=2 ("image")
    tdm_g1_t g1;
    g1[0] = (int)(1u << 16);                      // data_size=1 (2 bytes)
    g1[1] = (int)((tensor_w & 0xFFFFu) << 16);                    // tensor_dim0[15:0]
    g1[2] = (int)(((tensor_w >> 16) & 0xFFFFu) |
                  ((tensor_h & 0xFFFFu) << 16));                  // dim0 hi | dim1 lo
    g1[3] = (int)(((tensor_h >> 16) & 0xFFFFu) |
                  ((tile_w & 0xFFFFu) << 16));                    // dim1 hi | tile_dim0
    g1[4] = (int)(tile_h & 0xFFFFu);                              // tile_dim1 (tile_dim2=0)
    g1[5] = (int)stride;                                          // tensor_dim0_stride[31:0]
    g1[6] = 0;                                                    // stride0 hi | stride1 lo
    g1[7] = 0;                                                    // stride1 hi
    tdm_g2_t z4 = {0, 0, 0, 0};
#if __clang_major__ >= 23
    tdm_g1_t z8 = {0, 0, 0, 0, 0, 0, 0, 0};
    __builtin_amdgcn_tensor_load_to_lds(g0, g1, z4, z4, z8, 0);
#else
    __builtin_amdgcn_tensor_load_to_lds(g0, g1, z4, z4, 0);
#endif
}
#endif  // USE_TDM

// ---------------------------------------------------------------------------
// Kernel 1: scan visual_token_mask for first/last visual token
// ---------------------------------------------------------------------------
__global__ __launch_bounds__(256)
void k_scan_mask(const unsigned char* __restrict__ vm, float* __restrict__ scal) {
    __shared__ int smn[256], smx[256];
    int t = threadIdx.x;
    int mn = T_, mx = -1;
    for (int i = t; i < T_; i += 256)
        if (vm[i]) { mn = min(mn, i); mx = max(mx, i); }
    smn[t] = mn; smx[t] = mx;
    __syncthreads();
    for (int s = 128; s > 0; s >>= 1) {
        if (t < s) { smn[t] = min(smn[t], smn[t + s]); smx[t] = max(smx[t], smx[t + s]); }
        __syncthreads();
    }
    if (t == 0) {
        scal[0] = (float)smn[0];
        scal[1] = (float)smx[0];
        scal[2] = (smx[0] >= 0) ? 1.0f : 0.0f;
    }
}

// ---------------------------------------------------------------------------
// Kernel 2: fused dual-RoPE cos/sin table (combined position n + int(ids[n])).
// ---------------------------------------------------------------------------
__global__ __launch_bounds__(256)
void k_angles(const float* __restrict__ scal, float* __restrict__ cs, float* __restrict__ sn) {
    int idx = blockIdx.x * 256 + threadIdx.x;        // [0, T*64)
    int t = idx >> 6, d = idx & 63;
    float n = (float)t;
    float ids;
    if (scal[2] != 0.0f) {
        float vs = scal[0], ve = scal[1];
        float tt;
        if (n < vs)       tt = n;
        else if (n <= ve) tt = vs + floorf((n - vs) / FRAME_);
        else              tt = n - (ve - vs + 1.0f - floorf((ve - vs) / FRAME_));
        ids = n + GAMMA_ * tt;
    } else {
        ids = n;
    }
    float p = n + (float)((int)ids);
    float th = exp2f(-((float)(2 * d) * (1.0f / 128.0f)) * LOG2_10000_);
    float ang = p * th;
    cs[idx] = cosf(ang);
    sn[idx] = sinf(ang);
}

// ---------------------------------------------------------------------------
// Kernel 3: generic bf16 WMMA GEMM, double-buffered LDS (one barrier/k-step).
//   A: [M,K] row-major (fp32 converted on the fly, or raw bf16)
//   B: [N,K] row-major fp32 (computes A @ B^T)
//   OUT_MODE 0: fp32 per-head layout [N/128][M][128]   (QKV projections)
//   OUT_MODE 1: fp32 row-major [M,N]                   (final projection)
// ---------------------------------------------------------------------------
template <typename AT, int OUT_MODE>
__global__ __launch_bounds__(256)
void k_gemm(const AT* __restrict__ A, const float* __restrict__ Bw,
            float* __restrict__ Cout, int M, int N, int K) {
    __shared__ unsigned short At[2][128][32];
    __shared__ unsigned short Bt[2][128][32];

    const int tid  = threadIdx.x;
    const int wave = tid >> 5, lane = tid & 31;
    const int lm = lane & 15, lhi = lane >> 4;
    const int d8 = lhi * 8, d16 = lhi * 16;
    const int wm = wave >> 2, wn = wave & 3;           // 2 x 4 wave grid
    const int mb = blockIdx.y * 128, nb = blockIdx.x * 128;
    const int srow = tid >> 1, skh = (tid & 1) * 16;

    const AT*    arow = A  + (size_t)(mb + srow) * K + skh;
    const float* brow = Bw + (size_t)(nb + srow) * K + skh;

    v8f acc[4][2];
    #pragma unroll
    for (int i = 0; i < 4; ++i)
        #pragma unroll
        for (int j = 0; j < 2; ++j)
            #pragma unroll
            for (int e = 0; e < 8; ++e) acc[i][j][e] = 0.0f;

    // prologue: stage k-block 0 into buffer 0
    v8u alo, ahi, blo, bhi;
    ld16bf(arow, alo, ahi);
    ld16bf(brow, blo, bhi);
    *(v8u*)&At[0][srow][skh]     = alo;
    *(v8u*)&At[0][srow][skh + 8] = ahi;
    *(v8u*)&Bt[0][srow][skh]     = blo;
    *(v8u*)&Bt[0][srow][skh + 8] = bhi;

    for (int kb = 0; kb < K; kb += 32) {
        const int  p    = (kb >> 5) & 1;
        const bool more = (kb + 32) < K;
        __syncthreads();                       // buffer p published; prior reads done
        if (more) {
            ld16bf(arow + kb + 32, alo, ahi);
            ld16bf(brow + kb + 32, blo, bhi);
            __builtin_prefetch(arow + kb + 64, 0, 1);   // global_prefetch_b8
            __builtin_prefetch(brow + kb + 64, 0, 1);
        }

        v16bf afr[4], bfr[2];
        #pragma unroll
        for (int mt = 0; mt < 4; ++mt) {
            const unsigned short* pr = &At[p][wm * 64 + mt * 16 + lm][0];
            afr[mt] = bc16(*(const v8u*)(pr + d8), *(const v8u*)(pr + 16 + d8));
        }
        #pragma unroll
        for (int nt = 0; nt < 2; ++nt) {
            const unsigned short* pr = &Bt[p][wn * 32 + nt * 16 + lm][0];
            bfr[nt] = bc16(*(const v8u*)(pr + d8), *(const v8u*)(pr + 16 + d8));
        }
        #pragma unroll
        for (int mt = 0; mt < 4; ++mt)
            #pragma unroll
            for (int nt = 0; nt < 2; ++nt)
                acc[mt][nt] = wmma_bf16(afr[mt], bfr[nt], acc[mt][nt]);

        if (more) {
            *(v8u*)&At[p ^ 1][srow][skh]     = alo;
            *(v8u*)&At[p ^ 1][srow][skh + 8] = ahi;
            *(v8u*)&Bt[p ^ 1][srow][skh]     = blo;
            *(v8u*)&Bt[p ^ 1][srow][skh + 8] = bhi;
        }
    }

    #pragma unroll
    for (int mt = 0; mt < 4; ++mt)
        #pragma unroll
        for (int nt = 0; nt < 2; ++nt)
            #pragma unroll
            for (int v = 0; v < 8; ++v) {
                int m = mb + wm * 64 + mt * 16 + v + 8 * lhi;
                int n = nb + wn * 32 + nt * 16 + lm;
                float val = acc[mt][nt][v];
                if (OUT_MODE == 0)
                    Cout[((size_t)(n >> 7) * M + m) * DH_ + (n & 127)] = val;
                else
                    Cout[(size_t)m * N + n] = val;
            }
}

// ---------------------------------------------------------------------------
// Kernel 4: fused RoPE (combined angle) + bf16 conversion + V transpose.
// ---------------------------------------------------------------------------
__global__ __launch_bounds__(256)
void k_rope(const float* __restrict__ qf, const float* __restrict__ kf,
            const float* __restrict__ vf, const float* __restrict__ cs,
            const float* __restrict__ sn, unsigned short* __restrict__ qo,
            unsigned short* __restrict__ ko, unsigned short* __restrict__ vt) {
    size_t idx = (size_t)blockIdx.x * 256 + threadIdx.x;  // H*T*128
    int d = (int)(idx & 127);
    size_t ht = idx >> 7;                                  // h*T + t
    int t = (int)(ht & (T_ - 1));
    int h = (int)(ht >> 11);
    size_t base = ht * DH_;

    vt[((size_t)h * DH_ + d) * T_ + t] = f2bf(vf[base + d]);

    if (d < 64) {
        float c = cs[t * 64 + d], s = sn[t * 64 + d];
        float x1 = qf[base + d], x2 = qf[base + d + 64];
        qo[base + d]      = f2bf(x1 * c - x2 * s);
        qo[base + d + 64] = f2bf(x2 * c + x1 * s);
        x1 = kf[base + d]; x2 = kf[base + d + 64];
        ko[base + d]      = f2bf(x1 * c - x2 * s);
        ko[base + d + 64] = f2bf(x2 * c + x1 * s);
    }
}

// ---------------------------------------------------------------------------
// Kernel 5: causal flash attention.
//   Block = 4 waves = 64 q-rows of one head; K/V tiles (32 keys) staged into
//   double-buffered LDS by the Tensor Data Mover (wave 0 issues, TENSORcnt +
//   barrier synchronize), with a cooperative-copy fallback. Each wave owns a
//   16-row Q tile; waves past their causal diagonal skip compute uniformly.
// ---------------------------------------------------------------------------
__global__ __launch_bounds__(128)
void k_flash(const unsigned short* __restrict__ qbf,
             const unsigned short* __restrict__ kbf,
             const unsigned short* __restrict__ vtbf,
             unsigned short* __restrict__ attn) {
    __shared__ unsigned short Klds[2][32][128];   // [buf][key][dh]
    __shared__ unsigned short Vlds[2][128][32];   // [buf][dh][key]
    __shared__ unsigned short P[4][16][32];

    const int wave = threadIdx.x >> 5, lane = threadIdx.x & 31;
    const int lm = lane & 15, lhi = lane >> 4;
    const int d8 = lhi * 8, d16 = lhi * 16;
    const int h = blockIdx.x >> 5, qblk = blockIdx.x & 31;
    const int qb = qblk * 64 + wave * 16;
    const size_t hq = (size_t)h * T_ * DH_;
    const unsigned short* Kh = kbf + hq;
    const unsigned short* Vh = vtbf + (size_t)h * DH_ * T_;

    // Q A-fragments (16 rows x 128 dh = 4 fragments of 16x32)
    v16bf qfr[4];
    {
        const unsigned short* qrow = qbf + hq + (size_t)(qb + lm) * DH_;
        #pragma unroll
        for (int kc = 0; kc < 4; ++kc)
            qfr[kc] = bc16(*(const v8u*)(qrow + kc * 32 + d8),
                           *(const v8u*)(qrow + kc * 32 + 16 + d8));
    }

    v8f o[8];
    float mrun[8], lrun[8];
    #pragma unroll
    for (int nt = 0; nt < 8; ++nt)
        #pragma unroll
        for (int e = 0; e < 8; ++e) o[nt][e] = 0.0f;
    #pragma unroll
    for (int v = 0; v < 8; ++v) { mrun[v] = -1e30f; lrun[v] = 0.0f; }

    const int jd   = qb >> 5;         // this wave's diagonal 32-key block
    const int jmax = 2 * qblk + 1;    // block-wide trip count

    auto stage = [&](int j, int p) {
#if USE_TDM
        if (wave == 0) {
            tdm_load_2d((uint32_t)(uintptr_t)&Klds[p][0][0],
                        Kh + (size_t)(j * 32) * DH_,
                        /*tile_w*/ DH_, /*tile_h*/ 32, /*stride*/ DH_,
                        /*tensor_w*/ DH_, /*tensor_h*/ T_);
            tdm_load_2d((uint32_t)(uintptr_t)&Vlds[p][0][0],
                        Vh + (size_t)(j * 32),
                        /*tile_w*/ 32, /*tile_h*/ DH_, /*stride*/ T_,
                        /*tensor_w*/ T_, /*tensor_h*/ DH_);
        }
#else
        {   // cooperative copy, 128 threads
            int r  = threadIdx.x >> 2;          // key row 0..31
            int cq = (threadIdx.x & 3) * 32;    // 32-element chunk within row
            const v8u* src = (const v8u*)(Kh + ((size_t)(j * 32) + r) * DH_ + cq);
            v8u x0 = src[0], x1 = src[1], x2 = src[2], x3 = src[3];
            v8u* dst = (v8u*)&Klds[p][r][cq];
            dst[0] = x0; dst[1] = x1; dst[2] = x2; dst[3] = x3;
            const v8u* vs = (const v8u*)(Vh + (size_t)threadIdx.x * T_ + j * 32);
            v8u y0 = vs[0], y1 = vs[1];
            v8u* vd = (v8u*)&Vlds[p][threadIdx.x][0];
            vd[0] = y0; vd[1] = y1;
        }
#endif
    };

    stage(0, 0);
    for (int j = 0; j <= jmax; ++j) {
        const int p = j & 1;
        if (j < jmax) stage(j + 1, p ^ 1);
#if USE_TDM
        if (wave == 0) {
            if (j < jmax) __builtin_amdgcn_s_wait_tensorcnt(2);
            else          __builtin_amdgcn_s_wait_tensorcnt(0);
        }
#endif
        __syncthreads();                     // tile j published to all waves

        if (j <= jd) {                       // wave-uniform: EXEC stays all-ones
            const int kb0 = j * 32;

            v8f s[2];
            #pragma unroll
            for (int sub = 0; sub < 2; ++sub) {
                v8f a;
                #pragma unroll
                for (int e = 0; e < 8; ++e) a[e] = 0.0f;
                #pragma unroll
                for (int kc = 0; kc < 4; ++kc) {
                    const unsigned short* kr = &Klds[p][sub * 16 + lm][kc * 32 + d8];
                    a = wmma_bf16(qfr[kc],
                                  bc16(*(const v8u*)kr, *(const v8u*)(kr + 16)), a);
                }
                s[sub] = a;
            }

            const bool diag = (j == jd);
            #pragma unroll
            for (int v = 0; v < 8; ++v) {
                float a0 = s[0][v] * SCALE_;
                float a1 = s[1][v] * SCALE_;
                if (diag) {
                    int row = qb + v + 8 * lhi;
                    if (kb0 + lm > row)      a0 = -1e30f;
                    if (kb0 + 16 + lm > row) a1 = -1e30f;
                }
                float mx   = rmax16(fmaxf(a0, a1));
                float mnew = fmaxf(mrun[v], mx);
                float fac  = exp2f((mrun[v] - mnew) * LOG2E_);
                float p0   = exp2f((a0 - mnew) * LOG2E_);
                float p1   = exp2f((a1 - mnew) * LOG2E_);
                mrun[v] = mnew;
                lrun[v] = lrun[v] * fac + rsum16(p0 + p1);
                P[wave][v + 8 * lhi][lm]      = f2bf(p0);
                P[wave][v + 8 * lhi][16 + lm] = f2bf(p1);
                #pragma unroll
                for (int nt = 0; nt < 8; ++nt) o[nt][v] *= fac;
            }

            // per-wave LDS ordering: DS stores above land before reads below
            asm volatile("s_wait_dscnt 0x0" ::: "memory");

            v16bf pfr = bc16(*(const v8u*)&P[wave][lm][d8],
                             *(const v8u*)&P[wave][lm][16 + d8]);

            #pragma unroll
            for (int nt = 0; nt < 8; ++nt) {
                const unsigned short* vr = &Vlds[p][nt * 16 + lm][d8];
                o[nt] = wmma_bf16(pfr, bc16(*(const v8u*)vr, *(const v8u*)(vr + 16)),
                                  o[nt]);
            }
        }
        __syncthreads();                     // done reading buffer p
    }

    #pragma unroll
    for (int v = 0; v < 8; ++v) {
        float inv = 1.0f / lrun[v];
        int t = qb + v + 8 * lhi;
        size_t rowb = (size_t)t * C_ + h * DH_;
        #pragma unroll
        for (int nt = 0; nt < 8; ++nt)
            attn[rowb + nt * 16 + lm] = f2bf(o[nt][v] * inv);
    }
}

// ---------------------------------------------------------------------------
// Launcher. Workspace layout (~162 MB total):
//   [0,256)        scalar results (v_s, v_e, has_visual)
//   [4K, +512K]x2  cos/sin tables [T][64]
//   [2M, +32M]x3   fp32 Q/K/V in per-head layout [H][T][DH]
//   then 4x16M     bf16 Q, K, V^T, attn-out
// ---------------------------------------------------------------------------
extern "C" void kernel_launch(void* const* d_in, const int* in_sizes, int n_in,
                              void* d_out, int out_size, void* d_ws, size_t ws_size,
                              hipStream_t stream) {
    const float*         hs = (const float*)d_in[0];
    const unsigned char* vm = (const unsigned char*)d_in[1];  // numpy bool, 1 byte
    const float*         wq = (const float*)d_in[2];
    const float*         wk = (const float*)d_in[3];
    const float*         wv = (const float*)d_in[4];
    const float*         wo = (const float*)d_in[5];
    float* out = (float*)d_out;

    uint8_t* ws = (uint8_t*)d_ws;
    float* scal = (float*)(ws + 0);
    float* cs   = (float*)(ws + 4096);
    float* sn   = (float*)(ws + 4096 + (size_t)T_ * 64 * 4);
    size_t off = (size_t)2 << 20;
    float* qf = (float*)(ws + off); off += (size_t)T_ * C_ * 4;
    float* kf = (float*)(ws + off); off += (size_t)T_ * C_ * 4;
    float* vf = (float*)(ws + off); off += (size_t)T_ * C_ * 4;
    unsigned short* qb16 = (unsigned short*)(ws + off); off += (size_t)T_ * C_ * 2;
    unsigned short* kb16 = (unsigned short*)(ws + off); off += (size_t)T_ * C_ * 2;
    unsigned short* vt16 = (unsigned short*)(ws + off); off += (size_t)T_ * C_ * 2;
    unsigned short* at16 = (unsigned short*)(ws + off); off += (size_t)T_ * C_ * 2;

    k_scan_mask<<<1, 256, 0, stream>>>(vm, scal);
    k_angles<<<(T_ * 64) / 256, 256, 0, stream>>>(scal, cs, sn);

    dim3 gg(C_ / 128, T_ / 128);
    k_gemm<float, 0><<<gg, 256, 0, stream>>>(hs, wq, qf, T_, C_, C_);
    k_gemm<float, 0><<<gg, 256, 0, stream>>>(hs, wk, kf, T_, C_, C_);
    k_gemm<float, 0><<<gg, 256, 0, stream>>>(hs, wv, vf, T_, C_, C_);

    k_rope<<<(H_ * T_ * DH_) / 256, 256, 0, stream>>>(qf, kf, vf, cs, sn, qb16, kb16, vt16);

    k_flash<<<H_ * (T_ / 64), 128, 0, stream>>>(qb16, kb16, vt16, at16);

    k_gemm<unsigned short, 1><<<gg, 256, 0, stream>>>(at16, wo, out, T_, C_, C_);
}
